// LeWinTransformerBlock_3719441678493
// MI455X (gfx1250) — matrix operand order repaired and checked by reading, not compiled, not run on hardware
//
#include <hip/hip_runtime.h>
#include <hip/hip_bf16.h>
#include <math.h>

// ---------------------------------------------------------------------------
// LeWin transformer block for MI455X (gfx1250, wave32, WMMA bf16 16x16x32)
// ---------------------------------------------------------------------------

typedef __attribute__((ext_vector_type(16))) __bf16 v16bf;
typedef __attribute__((ext_vector_type(8)))  __bf16 v8bf;
typedef __attribute__((ext_vector_type(8)))  float  v8f;

#define WSZ    8
#define NTOK   64
#define CH     128
#define HEADS  4
#define HD     32
#define HID    8
#define IMG    256
#define NB     4

// LDS strides (elements), chosen for 16B alignment + bank-conflict freedom
#define SXST   136   // bf16, 64 x 136  (row stride 272B -> 4-bank shift/row)
#define SVST   72    // bf16, 128 x 72  (144B -> 36-dword shift, gcd(36,64)=4 ok)
#define SSST   68    // f32,  256 x 68  (272B rows)

__device__ __forceinline__ float gelu_exact(float v) {
    return 0.5f * v * (1.0f + erff(v * 0.70710678118654752f));
}

// Build a 16-element bf16 fragment from two 16B-aligned 8-element chunks
__device__ __forceinline__ v16bf ld_2x8(const __bf16* p0, const __bf16* p1) {
    v8bf a = *(const v8bf*)p0;
    v8bf b = *(const v8bf*)p1;
    v16bf r;
#pragma unroll
    for (int i = 0; i < 8; ++i) { r[i] = a[i]; r[i + 8] = b[i]; }
    return r;
}

// ---------------------------------------------------------------------------
// Per-(b,c) instance-norm statistics: mean + rsqrt(var+eps). 512 blocks.
// ---------------------------------------------------------------------------
__global__ __launch_bounds__(256) void stats_kernel(const float* __restrict__ src,
                                                    float* __restrict__ out) {
    const int bc = blockIdx.x;
    const int t  = threadIdx.x;
    const float4* p = (const float4*)(src + (size_t)bc * (IMG * IMG));
    float s = 0.f, ss = 0.f;
    for (int i = t; i < (IMG * IMG) / 4; i += 256) {
        float4 v = p[i];
        s  += v.x + v.y + v.z + v.w;
        ss += v.x * v.x + v.y * v.y + v.z * v.z + v.w * v.w;
    }
    __shared__ float rs[256], rss[256];
    rs[t] = s; rss[t] = ss;
    __syncthreads();
    for (int off = 128; off > 0; off >>= 1) {
        if (t < off) { rs[t] += rs[t + off]; rss[t] += rss[t + off]; }
        __syncthreads();
    }
    if (t == 0) {
        const float inv = 1.0f / (IMG * IMG);
        float mean = rs[0] * inv;
        float var  = rss[0] * inv - mean * mean;
        out[bc * 2 + 0] = mean;
        out[bc * 2 + 1] = rsqrtf(var + 1e-5f);
    }
}

// ---------------------------------------------------------------------------
// Convert qkv_w (384x128) and proj_w (128x128) f32 -> bf16 scratch copies
// ---------------------------------------------------------------------------
__global__ __launch_bounds__(256) void cvt_kernel(const float* __restrict__ qkv_w,
                                                  const float* __restrict__ proj_w,
                                                  __bf16* __restrict__ wq,
                                                  __bf16* __restrict__ wp) {
    int idx = blockIdx.x * 256 + threadIdx.x;   // 0 .. 65535
    if (idx < 3 * CH * CH) wq[idx] = (__bf16)qkv_w[idx];
    else                   wp[idx - 3 * CH * CH] = (__bf16)proj_w[idx - 3 * CH * CH];
}

// ---------------------------------------------------------------------------
// Fused window attention: norm+modulator -> QKV -> QK^T -> softmax(+rpb) ->
// P@V -> proj -> x_attn (NCHW f32).  One workgroup (8 wave32) per window.
// ---------------------------------------------------------------------------
__global__ __launch_bounds__(256) void attn_kernel(const float* __restrict__ x,
                                                   const float* __restrict__ stats,
                                                   const float* __restrict__ mod,
                                                   const __bf16* __restrict__ wq,
                                                   const float* __restrict__ qkv_b,
                                                   const __bf16* __restrict__ wp,
                                                   const float* __restrict__ proj_b,
                                                   const float* __restrict__ rpb,
                                                   float* __restrict__ xattn) {
    __shared__ __bf16 sX [NTOK * SXST];        // 17408 B : norm input, later O
    __shared__ __bf16 sQK[2 * NTOK * SXST];    // 34816 B : Q | K, later attOut f32
    __shared__ __bf16 sVT[CH * SVST];          // 18432 B : V transposed [c][n]
    __shared__ float  sS [HEADS * NTOK * SSST];// 69632 B : scores, P bf16 in-place
    __shared__ float  sRPB[225 * HEADS];       // 3600 B
    __shared__ float  sMu[CH], sRs[CH];

    const int t     = threadIdx.x;
    const int wg    = blockIdx.x;          // 0..4095
    const int b     = wg >> 10;
    const int wy    = (wg >> 5) & 31;
    const int wx    = wg & 31;
    const int y0    = wy * WSZ, x0 = wx * WSZ;
    const int w     = t >> 5;              // wave id 0..7
    const int lane  = t & 31;
    const int l15   = lane & 15;
    const int lhalf = lane >> 4;

    __bf16* sQ = sQK;
    __bf16* sK = sQK + NTOK * SXST;

    // ---- phase 0: stage stats, rpb; load window, normalize, + modulator ----
    if (t < CH) {
        sMu[t] = stats[(b * CH + t) * 2 + 0];
        sRs[t] = stats[(b * CH + t) * 2 + 1];
    }
    for (int i = t; i < 225 * HEADS; i += 256) sRPB[i] = rpb[i];
    __syncthreads();

#pragma unroll
    for (int i = 0; i < 4; ++i) {
        int id = t + i * 256;              // 0..1023 -> (c, iy)
        int c  = id >> 3, iy = id & 7;
        const float* gp = x + ((((size_t)b * CH + c) * IMG + (y0 + iy)) * IMG + x0);
        float4 v0 = *(const float4*)gp;
        float4 v1 = *(const float4*)(gp + 4);
        const float* mp = mod + (c * WSZ + iy) * WSZ;
        float4 m0 = *(const float4*)mp;
        float4 m1 = *(const float4*)(mp + 4);
        float mu = sMu[c], rs = sRs[c];
        int nb = iy * 8;
        sX[(nb + 0) * SXST + c] = (__bf16)((v0.x - mu) * rs + m0.x);
        sX[(nb + 1) * SXST + c] = (__bf16)((v0.y - mu) * rs + m0.y);
        sX[(nb + 2) * SXST + c] = (__bf16)((v0.z - mu) * rs + m0.z);
        sX[(nb + 3) * SXST + c] = (__bf16)((v0.w - mu) * rs + m0.w);
        sX[(nb + 4) * SXST + c] = (__bf16)((v1.x - mu) * rs + m1.x);
        sX[(nb + 5) * SXST + c] = (__bf16)((v1.y - mu) * rs + m1.y);
        sX[(nb + 6) * SXST + c] = (__bf16)((v1.z - mu) * rs + m1.z);
        sX[(nb + 7) * SXST + c] = (__bf16)((v1.w - mu) * rs + m1.w);
    }
    __syncthreads();

    // ---- phase 1: QKV = sX @ Wqkv^T + b  (64x384, tiles 4x24) ----
    // Split into Q / K / V segments so each store target is compile-time
    // static (no per-element EXEC-mask branching).
    {
        // ---- Q segment: otile = w, o_global = w*16 + l15 ----
        const int oq = w * 16 + l15;
        v16bf bfr[4];
#pragma unroll
        for (int k = 0; k < 4; ++k)
            bfr[k] = *(const v16bf*)(wq + oq * CH + k * 32 + lhalf * 16);
        float bias = qkv_b[oq];
#pragma unroll
        for (int nt = 0; nt < 4; ++nt) {
            v8f acc = {};
#pragma unroll
            for (int k = 0; k < 4; ++k) {
                const __bf16* ap = &sX[(nt * 16 + l15) * SXST + k * 32 + lhalf * 16];
                acc = __builtin_amdgcn_wmma_f32_16x16x32_bf16(
                          false, ld_2x8(ap, ap + 8), false, bfr[k], (short)0, acc, false, false);
            }
            int nbase = nt * 16 + lhalf * 8;
#pragma unroll
            for (int r = 0; r < 8; ++r)
                sQ[(nbase + r) * SXST + oq] = (__bf16)(acc[r] + bias);
        }
    }
    {
        // ---- K segment: otile = w+8, o_global = 128 + w*16 + l15 ----
        const int ok = w * 16 + l15;           // local K column
        v16bf bfr[4];
#pragma unroll
        for (int k = 0; k < 4; ++k)
            bfr[k] = *(const v16bf*)(wq + (CH + ok) * CH + k * 32 + lhalf * 16);
        float bias = qkv_b[CH + ok];
#pragma unroll
        for (int nt = 0; nt < 4; ++nt) {
            v8f acc = {};
#pragma unroll
            for (int k = 0; k < 4; ++k) {
                const __bf16* ap = &sX[(nt * 16 + l15) * SXST + k * 32 + lhalf * 16];
                acc = __builtin_amdgcn_wmma_f32_16x16x32_bf16(
                          false, ld_2x8(ap, ap + 8), false, bfr[k], (short)0, acc, false, false);
            }
            int nbase = nt * 16 + lhalf * 8;
#pragma unroll
            for (int r = 0; r < 8; ++r)
                sK[(nbase + r) * SXST + ok] = (__bf16)(acc[r] + bias);
        }
    }
    {
        // ---- V segment: otile = w+16, o_global = 256 + w*16 + l15 ----
        // vT rows are channel-major -> the 8 results are contiguous in n:
        // single 16B LDS store per tile.
        const int ov = w * 16 + l15;           // local V channel (row of sVT)
        v16bf bfr[4];
#pragma unroll
        for (int k = 0; k < 4; ++k)
            bfr[k] = *(const v16bf*)(wq + (2 * CH + ov) * CH + k * 32 + lhalf * 16);
        float bias = qkv_b[2 * CH + ov];
#pragma unroll
        for (int nt = 0; nt < 4; ++nt) {
            v8f acc = {};
#pragma unroll
            for (int k = 0; k < 4; ++k) {
                const __bf16* ap = &sX[(nt * 16 + l15) * SXST + k * 32 + lhalf * 16];
                acc = __builtin_amdgcn_wmma_f32_16x16x32_bf16(
                          false, ld_2x8(ap, ap + 8), false, bfr[k], (short)0, acc, false, false);
            }
            int nbase = nt * 16 + lhalf * 8;
            v8bf vs;
#pragma unroll
            for (int r = 0; r < 8; ++r) vs[r] = (__bf16)(acc[r] + bias);
            *(v8bf*)&sVT[ov * SVST + nbase] = vs;
        }
    }
    __syncthreads();

    // ---- phase 2: S[h] = Q_h @ K_h^T  (K=32=head_dim, one WMMA/tile) ----
    {
        int h = w & 3, nt0 = (w >> 2) * 2;
#pragma unroll
        for (int ni = 0; ni < 2; ++ni) {
            int nt = nt0 + ni;
            const __bf16* ap = &sQ[(nt * 16 + l15) * SXST + h * HD + lhalf * 16];
            v16bf afr = ld_2x8(ap, ap + 8);
#pragma unroll
            for (int mt = 0; mt < 4; ++mt) {
                const __bf16* bp = &sK[(mt * 16 + l15) * SXST + h * HD + lhalf * 16];
                v16bf bfr = ld_2x8(bp, bp + 8);
                v8f acc = {};
                acc = __builtin_amdgcn_wmma_f32_16x16x32_bf16(
                          false, afr, false, bfr, (short)0, acc, false, false);
                int nbase = nt * 16 + lhalf * 8;
#pragma unroll
                for (int r = 0; r < 8; ++r)
                    sS[(h * NTOK + nbase + r) * SSST + mt * 16 + l15] = acc[r];
            }
        }
    }
    __syncthreads();

    // ---- phase 3: row softmax with scale + relative position bias ----
    {
        int h = t >> 6, n = t & 63;
        int iy = n >> 3, ix = n & 7;
        float* srow = &sS[(h * NTOK + n) * SSST];
        float vals[NTOK];
        float mx = -3.4e38f;
#pragma unroll
        for (int m = 0; m < NTOK; ++m) {
            int jy = m >> 3, jx = m & 7;
            int idx = (iy - jy + 7) * 15 + (ix - jx + 7);
            float v = srow[m] * 0.17677669529663687f + sRPB[idx * HEADS + h];
            vals[m] = v;
            mx = fmaxf(mx, v);
        }
        float sum = 0.f;
#pragma unroll
        for (int m = 0; m < NTOK; ++m) { float e = __expf(vals[m] - mx); vals[m] = e; sum += e; }
        float inv = 1.0f / sum;
        asm volatile("" ::: "memory");   // order f32 reads before bf16 in-place writes
        __bf16* prow = reinterpret_cast<__bf16*>(srow);
#pragma unroll
        for (int m = 0; m < NTOK; ++m) prow[m] = (__bf16)(vals[m] * inv);
    }
    __syncthreads();

    // ---- phase 4: O = P @ V  -> sX (bf16 row-major [n][c]) ----
#pragma unroll
    for (int j = 0; j < 4; ++j) {
        int tid2 = w * 4 + j;              // 0..31
        int h = tid2 >> 3, nt = (tid2 >> 1) & 3, dt = tid2 & 1;
        v8f acc = {};
#pragma unroll
        for (int k = 0; k < 2; ++k) {
            const __bf16* prow =
                reinterpret_cast<const __bf16*>(&sS[(h * NTOK + nt * 16 + l15) * SSST]);
            const __bf16* ap = prow + k * 32 + lhalf * 16;
            v16bf afr = ld_2x8(ap, ap + 8);
            const __bf16* bp = &sVT[(h * HD + dt * 16 + l15) * SVST + k * 32 + lhalf * 16];
            v16bf bfr = ld_2x8(bp, bp + 8);
            acc = __builtin_amdgcn_wmma_f32_16x16x32_bf16(
                      false, afr, false, bfr, (short)0, acc, false, false);
        }
        int nbase = nt * 16 + lhalf * 8;
        int o = h * HD + dt * 16 + l15;
#pragma unroll
        for (int r = 0; r < 8; ++r)
            sX[(nbase + r) * SXST + o] = (__bf16)acc[r];
    }
    __syncthreads();

    // ---- phase 5: out = O @ Wproj^T + b -> attOut[c][n] f32 (aliases sQK) ----
    {
        float* attOut = reinterpret_cast<float*>(sQK);
        int o = w * 16 + l15;              // 8 waves x 16 = 128 channels
        v16bf bfr[4];
#pragma unroll
        for (int k = 0; k < 4; ++k)
            bfr[k] = *(const v16bf*)(wp + o * CH + k * 32 + lhalf * 16);
        float pb = proj_b[o];
#pragma unroll
        for (int nt = 0; nt < 4; ++nt) {
            v8f acc = {};
#pragma unroll
            for (int k = 0; k < 4; ++k) {
                const __bf16* ap = &sX[(nt * 16 + l15) * SXST + k * 32 + lhalf * 16];
                acc = __builtin_amdgcn_wmma_f32_16x16x32_bf16(
                          false, ld_2x8(ap, ap + 8), false, bfr[k], (short)0, acc, false, false);
            }
            int nbase = nt * 16 + lhalf * 8;
#pragma unroll
            for (int r = 0; r < 8; ++r)
                attOut[o * SSST + nbase + r] = acc[r] + pb;
        }
    }
    __syncthreads();

    // ---- phase 6: coalesced NCHW store of x_attn ----
    {
        const float* attOut = reinterpret_cast<const float*>(sQK);
#pragma unroll
        for (int i = 0; i < 4; ++i) {
            int id = t + i * 256;
            int c = id >> 3, iy = id & 7;
            const float* sp = &attOut[c * SSST + iy * 8];
            float4 a = *(const float4*)sp;
            float4 bb = *(const float4*)(sp + 4);
            float* gp = xattn + ((((size_t)b * CH + c) * IMG + (y0 + iy)) * IMG + x0);
            *(float4*)gp = a;
            *(float4*)(gp + 4) = bb;
        }
    }
}

// ---------------------------------------------------------------------------
// LeFF part 1: h1 = GELU( conv1x1(instnorm(x_attn)) )   (128 -> 8 channels)
// One block per (b, y) row; coalesced channel-plane access.
// ---------------------------------------------------------------------------
__global__ __launch_bounds__(256) void leff1_kernel(const float* __restrict__ xattn,
                                                    const float* __restrict__ stats,
                                                    const float* __restrict__ w1,
                                                    const float* __restrict__ b1,
                                                    float* __restrict__ h1) {
    __shared__ float smu[CH], srs[CH];
    const int t = threadIdx.x;
    const int b = blockIdx.x >> 8;
    const int y = blockIdx.x & 255;
    if (t < CH) {
        smu[t] = stats[(b * CH + t) * 2 + 0];
        srs[t] = stats[(b * CH + t) * 2 + 1];
    }
    __syncthreads();

    const float* src = xattn + (size_t)b * CH * IMG * IMG + y * IMG + t;
    float acc[HID];
#pragma unroll
    for (int o = 0; o < HID; ++o) acc[o] = 0.f;
    for (int c = 0; c < CH; ++c) {
        float xv = (src[(size_t)c * IMG * IMG] - smu[c]) * srs[c];
#pragma unroll
        for (int o = 0; o < HID; ++o)
            acc[o] = fmaf(w1[o * CH + c], xv, acc[o]);   // uniform -> scalar loads
    }
#pragma unroll
    for (int o = 0; o < HID; ++o) {
        float v = gelu_exact(acc[o] + b1[o]);
        h1[((size_t)b * HID + o) * IMG * IMG + y * IMG + t] = v;
    }
}

// ---------------------------------------------------------------------------
// LeFF part 2: GELU(dwconv3x3(h1)) -> conv1x1 (8 -> 128) + b2 + residual x
// ---------------------------------------------------------------------------
__global__ __launch_bounds__(256) void leff2_kernel(const float* __restrict__ h1,
                                                    const float* __restrict__ dw_w,
                                                    const float* __restrict__ dw_b,
                                                    const float* __restrict__ w2,
                                                    const float* __restrict__ b2,
                                                    const float* __restrict__ x,
                                                    float* __restrict__ out) {
    const int t = threadIdx.x;   // pixel x coordinate
    const int b = blockIdx.x >> 8;
    const int y = blockIdx.x & 255;

    float h2[HID];
#pragma unroll
    for (int o = 0; o < HID; ++o) {
        const float* hp = h1 + ((size_t)b * HID + o) * IMG * IMG;
        float acc = 0.f;
#pragma unroll
        for (int dy = -1; dy <= 1; ++dy) {
            int yy = y + dy;
            if (yy < 0 || yy >= IMG) continue;
#pragma unroll
            for (int dx = -1; dx <= 1; ++dx) {
                int xx = t + dx;
                if (xx < 0 || xx >= IMG) continue;
                acc = fmaf(dw_w[o * 9 + (dy + 1) * 3 + (dx + 1)], hp[yy * IMG + xx], acc);
            }
        }
        h2[o] = gelu_exact(acc + dw_b[o]);
    }

    const float* xr = x + (size_t)b * CH * IMG * IMG + y * IMG + t;
    float* op = out + (size_t)b * CH * IMG * IMG + y * IMG + t;
    for (int c = 0; c < CH; ++c) {
        float s = b2[c];
#pragma unroll
        for (int o = 0; o < HID; ++o)
            s = fmaf(w2[c * HID + o], h2[o], s);
        op[(size_t)c * IMG * IMG] = s + xr[(size_t)c * IMG * IMG];
    }
}

// ---------------------------------------------------------------------------
// Launch
// ---------------------------------------------------------------------------
extern "C" void kernel_launch(void* const* d_in, const int* in_sizes, int n_in,
                              void* d_out, int out_size, void* d_ws, size_t ws_size,
                              hipStream_t stream) {
    const float* x         = (const float*)d_in[0];
    const float* modulator = (const float*)d_in[1];
    const float* qkv_w     = (const float*)d_in[2];
    const float* qkv_b     = (const float*)d_in[3];
    const float* proj_w    = (const float*)d_in[4];
    const float* proj_b    = (const float*)d_in[5];
    const float* rpb       = (const float*)d_in[6];
    const float* w1        = (const float*)d_in[7];
    const float* b1        = (const float*)d_in[8];
    const float* dw_w      = (const float*)d_in[9];
    const float* dw_b      = (const float*)d_in[10];
    const float* w2        = (const float*)d_in[11];
    const float* b2        = (const float*)d_in[12];
    float* out = (float*)d_out;

    // Workspace layout (~8.5 MB):
    char* ws = (char*)d_ws;
    float*  stats1 = (float*)(ws + 0);            // 512*2 f32
    float*  stats2 = (float*)(ws + 4096);         // 512*2 f32
    __bf16* wqb    = (__bf16*)(ws + 8192);        // 384*128 bf16
    __bf16* wpb    = (__bf16*)(ws + 8192 + 98304);// 128*128 bf16
    float*  h1buf  = (float*)(ws + 139264);       // 4*8*256*256 f32

    // 1) instance-norm stats of x
    stats_kernel<<<NB * CH, 256, 0, stream>>>(x, stats1);
    // 2) bf16 weight copies
    cvt_kernel<<<256, 256, 0, stream>>>(qkv_w, proj_w, wqb, wpb);
    // 3) fused window attention -> x_attn (stored in d_out temporarily)
    attn_kernel<<<NB * 32 * 32, 256, 0, stream>>>(x, stats1, modulator, wqb, qkv_b,
                                                  wpb, proj_b, rpb, out);
    // 4) instance-norm stats of x_attn
    stats_kernel<<<NB * CH, 256, 0, stream>>>(out, stats2);
    // 5) LeFF conv1 + GELU
    leff1_kernel<<<NB * IMG, 256, 0, stream>>>(out, stats2, w1, b1, h1buf);
    // 6) dwconv + GELU + conv2 + residual -> final output (overwrites d_out)
    leff2_kernel<<<NB * IMG, 256, 0, stream>>>(h1buf, dw_w, dw_b, w2, b2, x, out);
}
